// graph_constructor_71683004171047
// MI455X (gfx1250) — compile-verified, matrix operand order checked by reading, not run on hardware
//
#include <hip/hip_runtime.h>
#include <hip/hip_bf16.h>

typedef __bf16 bf16_t;
typedef __attribute__((ext_vector_type(16))) __bf16 v16bf;
typedef __attribute__((ext_vector_type(8)))  __bf16 v8bf;
typedef __attribute__((ext_vector_type(8)))  float  v8f;

#define N_VARS  2048
#define D_GRAPH 256
#define NUM_ADJ 7
#define NBINS   1024
#define TOPK    ((unsigned long long)N_VARS * N_VARS / 2)

#define WMMA_BF16(A, B, C) \
  __builtin_amdgcn_wmma_f32_16x16x32_bf16(false, (A), false, (B), (short)0, (C), false, false)

// CDNA5 has a hardware transcendental V_TANH_F32 (TRANS unit, co-executes
// with WMMA). Prefer the builtin; fall back to inline asm.
static __device__ inline float fast_tanh(float x) {
#if __has_builtin(__builtin_amdgcn_tanhf)
  return __builtin_amdgcn_tanhf(x);
#elif __has_builtin(__builtin_amdgcn_tanh_f32)
  return __builtin_amdgcn_tanh_f32(x);
#else
  float r;
  asm volatile("v_tanh_f32 %0, %1\n\ts_nop 1" : "=v"(r) : "v"(x));
  return r;
#endif
}

static __device__ inline void wait_async0() {
#if __has_builtin(__builtin_amdgcn_s_wait_asynccnt)
  __builtin_amdgcn_s_wait_asynccnt(0);
#else
  asm volatile("s_wait_asynccnt 0" ::: "memory");
#endif
}

// Issue GLOBAL_LOAD_ASYNC_TO_LDS_B128 (GVS mode): 64-bit SGPR base + 32-bit
// VGPR byte offset -> 16 bytes written directly to LDS (tracked by ASYNCcnt).
static __device__ inline void async_copy_b128(unsigned lds_addr, unsigned gbyte_off,
                                              const void* sbase) {
  asm volatile("global_load_async_to_lds_b128 %0, %1, %2"
               :: "v"(lds_addr), "v"(gbyte_off),
                  "s"((unsigned long long)(size_t)sbase)
               : "memory");
}

// Per-lane bf16 fragment from a row-major panel with given row stride (elems).
// Lane L (lane16 = L%16 row-in-tile; hi = L/16):
//  elements 0..7  = K[hi*8 + 0..7], elements 8..15 = K[hi*8 + 16..23]
static __device__ inline v16bf load_frag_bf16(const bf16_t* rowbase, int hi) {
  union { v16bf v; v8bf h[2]; } u;
  const v8bf* p = (const v8bf*)(rowbase + hi * 8);
  u.h[0] = p[0];   // +0..7
  u.h[1] = p[2];   // +16..23
  return u.v;
}

// Same fragment pattern from fp32 source with on-the-fly bf16 conversion.
static __device__ inline v16bf load_frag_f32(const float* rowbase, int kb, int hi) {
  const float* p = rowbase + kb + hi * 8;
  v16bf v;
#pragma unroll
  for (int t = 0; t < 8; ++t) {
    v[t]     = (bf16_t)p[t];
    v[t + 8] = (bf16_t)p[t + 16];
  }
  return v;
}

// ---------------------------------------------------------------------------
// Kernel 1: nv = tanh(e @ W^T + b), bf16 output. One 32x32 tile per wave.
// Waves: NUM_ADJ * (2048/32) * (256/32) = 3584 -> 448 blocks of 8 waves.
// ---------------------------------------------------------------------------
__global__ void __launch_bounds__(256) nv_gemm_kernel(const float* __restrict__ emb,
                                                      const float* __restrict__ W,
                                                      const float* __restrict__ bias,
                                                      bf16_t* __restrict__ nv) {
  const int wave   = blockIdx.x * 8 + (threadIdx.x >> 5);
  const int lane   = threadIdx.x & 31;
  const int lane16 = lane & 15;
  const int hi     = lane >> 4;
  const int g  = wave / 512;
  const int t  = wave % 512;
  const int tm = (t >> 3) * 32;
  const int tn = (t & 7) * 32;

  const float* e   = emb  + (size_t)g * N_VARS * D_GRAPH;
  const float* Wg  = W    + (size_t)g * D_GRAPH * D_GRAPH;
  const float* bg  = bias + (size_t)g * D_GRAPH;
  bf16_t*      nvg = nv   + (size_t)g * N_VARS * D_GRAPH;

  v8f c[2][2] = {};
#pragma unroll 1
  for (int kb = 0; kb < D_GRAPH; kb += 32) {
    v16bf a0 = load_frag_f32(e  + (size_t)(tm + lane16)      * D_GRAPH, kb, hi);
    v16bf a1 = load_frag_f32(e  + (size_t)(tm + 16 + lane16) * D_GRAPH, kb, hi);
    v16bf b0 = load_frag_f32(Wg + (size_t)(tn + lane16)      * D_GRAPH, kb, hi);
    v16bf b1 = load_frag_f32(Wg + (size_t)(tn + 16 + lane16) * D_GRAPH, kb, hi);
    c[0][0] = WMMA_BF16(a0, b0, c[0][0]);
    c[0][1] = WMMA_BF16(a0, b1, c[0][1]);
    c[1][0] = WMMA_BF16(a1, b0, c[1][0]);
    c[1][1] = WMMA_BF16(a1, b1, c[1][1]);
  }
#pragma unroll
  for (int mi = 0; mi < 2; ++mi)
#pragma unroll
    for (int ni = 0; ni < 2; ++ni)
#pragma unroll
      for (int r = 0; r < 8; ++r) {
        int m = tm + mi * 16 + hi * 8 + r;
        int n = tn + ni * 16 + lane16;
        float val = fast_tanh(c[mi][ni][r] + bg[n]);
        nvg[(size_t)m * D_GRAPH + n] = (bf16_t)val;
      }
}

// ---------------------------------------------------------------------------
// Kernel 2: P = relu(tanh((1-prop)*init + prop * nv1 @ nv2^T)), diag zeroed.
// Block = 128x256 macro-tile, 8 waves (2x4) of 64x64. K panels (A:128x32,
// B:256x32 bf16) double-buffered in LDS via async global->LDS copies.
// Blocks: 7 * (2048/128) * (2048/256) = 896.
// ---------------------------------------------------------------------------
__global__ void __launch_bounds__(256) adj_gemm_kernel(const bf16_t* __restrict__ nv1,
                                                       const bf16_t* __restrict__ nv2,
                                                       const float* __restrict__ init_adj,
                                                       const int* __restrict__ epoch_p,
                                                       float* __restrict__ P) {
  __shared__ bf16_t Abuf[2][128 * 32];   // 2 x 8 KB
  __shared__ bf16_t Bbuf[2][256 * 32];   // 2 x 16 KB

  const int tid    = threadIdx.x;
  const int lane   = tid & 31;
  const int lane16 = lane & 15;
  const int hi     = lane >> 4;
  const int wid    = tid >> 5;
  const int warp_m = wid >> 2;            // 0..1
  const int warp_n = wid & 3;             // 0..3

  const int g  = blockIdx.x >> 7;         // 128 macro-tiles per adjacency
  const int t  = blockIdx.x & 127;
  const int block_tm = (t >> 3) * 128;    // 16 M macro-tiles
  const int block_tn = (t & 7) * 256;     // 8  N macro-tiles

  const bf16_t* n1 = nv1 + (size_t)g * N_VARS * D_GRAPH;
  const bf16_t* n2 = nv2 + (size_t)g * N_VARS * D_GRAPH;
  float*        Pg = P   + (size_t)g * N_VARS * N_VARS;

  // Stage one K panel (32 wide) of A and B into LDS buffer `buf`.
  auto stage = [&](int step, int buf) {
    const int kb = step * 32;
    // A: 128 rows x 32 cols = 512 x 16B chunks; 2 per thread
#pragma unroll
    for (int s = 0; s < 2; ++s) {
      int idx = tid + s * 256;
      int row = idx >> 2, chunk = idx & 3;
      unsigned goff = (unsigned)(((block_tm + row) * D_GRAPH + kb + chunk * 8) * 2);
      unsigned lds  = (unsigned)(size_t)&Abuf[buf][row * 32 + chunk * 8];
      async_copy_b128(lds, goff, n1);
    }
    // B: 256 rows x 32 cols = 1024 x 16B chunks; 4 per thread
#pragma unroll
    for (int s = 0; s < 4; ++s) {
      int idx = tid + s * 256;
      int row = idx >> 2, chunk = idx & 3;
      unsigned goff = (unsigned)(((block_tn + row) * D_GRAPH + kb + chunk * 8) * 2);
      unsigned lds  = (unsigned)(size_t)&Bbuf[buf][row * 32 + chunk * 8];
      async_copy_b128(lds, goff, n2);
    }
  };

  v8f c[4][4] = {};
  stage(0, 0);

#pragma unroll 1
  for (int step = 0; step < D_GRAPH / 32; ++step) {
    const int buf = step & 1;
    wait_async0();           // my async writes for `buf` are in LDS
    __syncthreads();         // everyone's are
    if (step + 1 < D_GRAPH / 32) stage(step + 1, buf ^ 1);

    v16bf a[4], b[4];
#pragma unroll
    for (int i = 0; i < 4; ++i)
      a[i] = load_frag_bf16(&Abuf[buf][(warp_m * 64 + i * 16 + lane16) * 32], hi);
#pragma unroll
    for (int j = 0; j < 4; ++j)
      b[j] = load_frag_bf16(&Bbuf[buf][(warp_n * 64 + j * 16 + lane16) * 32], hi);
#pragma unroll
    for (int i = 0; i < 4; ++i)
#pragma unroll
      for (int j = 0; j < 4; ++j)
        c[i][j] = WMMA_BF16(a[i], b[j], c[i][j]);
  }

  const float prop = fminf((float)epoch_p[0] / 5.0f, 0.9f);
  const float p0   = 1.0f - prop;
  const int tm = block_tm + warp_m * 64;
  const int tn = block_tn + warp_n * 64;
#pragma unroll
  for (int i = 0; i < 4; ++i)
#pragma unroll
    for (int j = 0; j < 4; ++j)
#pragma unroll
      for (int r = 0; r < 8; ++r) {
        int m = tm + i * 16 + hi * 8 + r;
        int n = tn + j * 16 + lane16;
        float a0 = p0 * init_adj[(size_t)m * N_VARS + n] + prop * c[i][j][r];
        float v = fast_tanh(a0);
        v = v > 0.0f ? v : 0.0f;   // relu
        if (m == n) v = 0.0f;      // remove diagonal
        Pg[(size_t)m * N_VARS + n] = v;
      }
}

// ---------------------------------------------------------------------------
// Histogram over P values in [0,1) for the top-k threshold. LDS-privatized.
// ---------------------------------------------------------------------------
__global__ void zero_hist_kernel(unsigned int* hist) {
  int idx = blockIdx.x * blockDim.x + threadIdx.x;
  if (idx < NUM_ADJ * NBINS) hist[idx] = 0u;
}

__global__ void __launch_bounds__(256) hist_kernel(const float* __restrict__ P,
                                                   unsigned int* __restrict__ hist) {
  __shared__ unsigned int lh[NBINS];
  const int g = blockIdx.y;
  for (int i = threadIdx.x; i < NBINS; i += 256) lh[i] = 0u;
  __syncthreads();
  const float* Pg = P + (size_t)g * N_VARS * N_VARS;
  size_t base = (size_t)blockIdx.x * (256 * 8) + threadIdx.x;
#pragma unroll
  for (int s = 0; s < 8; ++s) {
    float v = Pg[base + (size_t)s * 256];
    int b = (int)(v * (float)NBINS);
    b = b < 0 ? 0 : (b > NBINS - 1 ? NBINS - 1 : b);
    atomicAdd(&lh[b], 1u);
  }
  __syncthreads();
  for (int i = threadIdx.x; i < NBINS; i += 256)
    if (lh[i]) atomicAdd(&hist[g * NBINS + i], lh[i]);
}

// Suffix-scan from the top bin: threshold = lower edge of the bin where the
// running count of larger values first reaches k (= n*n/2).
__global__ void thresh_kernel(const unsigned int* __restrict__ hist,
                              float* __restrict__ thr) {
  const int g = blockIdx.x;
  if (threadIdx.x == 0) {
    unsigned long long cum = 0;
    int bsel = 0;
    for (int b = NBINS - 1; b >= 0; --b) {
      cum += hist[g * NBINS + b];
      if (cum >= TOPK) { bsel = b; break; }
    }
    thr[g] = (float)bsel / (float)NBINS;
  }
}

// ---------------------------------------------------------------------------
// Finalize: gather by time_indices, apply threshold mask, add eye, normalize.
// One block per output row (b, i); 2048 elements, 8 per thread.
// ---------------------------------------------------------------------------
__global__ void __launch_bounds__(256) finalize_kernel(const float* __restrict__ P,
                                                       const float* __restrict__ thr,
                                                       const int* __restrict__ time_indices,
                                                       float* __restrict__ out) {
  __shared__ float red[256];
  const int b = blockIdx.y;
  const int i = blockIdx.x;
  const int g = time_indices[b];
  const float tv = thr[g];
  const float* row = P + ((size_t)g * N_VARS + i) * N_VARS;

  float vals[8];
  float partial = 0.0f;
#pragma unroll
  for (int s = 0; s < 8; ++s) {
    int j = threadIdx.x + s * 256;
    float v = row[j];
    v = (v >= tv) ? v : 0.0f;
    vals[s] = v;
    partial += v;
  }
  red[threadIdx.x] = partial;
  __syncthreads();
  for (int off = 128; off > 0; off >>= 1) {
    if (threadIdx.x < off) red[threadIdx.x] += red[threadIdx.x + off];
    __syncthreads();
  }
  const float inv = 1.0f / (red[0] + 1.0f);   // +1 for the eye diagonal
  float* orow = out + ((size_t)b * N_VARS + i) * N_VARS;
#pragma unroll
  for (int s = 0; s < 8; ++s) {
    int j = threadIdx.x + s * 256;
    orow[j] = (j == i) ? inv : vals[s] * inv;
  }
}

extern "C" void kernel_launch(void* const* d_in, const int* in_sizes, int n_in,
                              void* d_out, int out_size, void* d_ws, size_t ws_size,
                              hipStream_t stream) {
  (void)in_sizes; (void)n_in; (void)out_size; (void)ws_size;
  const float* init_adj     = (const float*)d_in[0];
  const float* emb1         = (const float*)d_in[1];
  const float* emb2         = (const float*)d_in[2];
  const float* w1           = (const float*)d_in[3];
  const float* b1           = (const float*)d_in[4];
  const float* w2           = (const float*)d_in[5];
  const float* b2           = (const float*)d_in[6];
  const int*   time_indices = (const int*)d_in[7];
  const int*   epoch        = (const int*)d_in[8];
  float* out = (float*)d_out;

  // Workspace layout:
  //   [0,     28672) : histograms (7 x 1024 u32)
  //   [32768, 32796) : thresholds (7 f32)
  //   [65536, ...  ) : nv1 bf16 [7][2048][256], nv2 bf16 [7][2048][256],
  //                    P f32 [7][2048][2048]
  char* ws = (char*)d_ws;
  unsigned int* hist = (unsigned int*)ws;
  float*        thr  = (float*)(ws + 32768);
  bf16_t*       nv1  = (bf16_t*)(ws + 65536);
  bf16_t*       nv2  = (bf16_t*)(ws + 65536 + (size_t)NUM_ADJ * N_VARS * D_GRAPH * 2);
  float*        P    = (float*) (ws + 65536 + (size_t)2 * NUM_ADJ * N_VARS * D_GRAPH * 2);

  zero_hist_kernel<<<(NUM_ADJ * NBINS + 255) / 256, 256, 0, stream>>>(hist);

  nv_gemm_kernel<<<448, 256, 0, stream>>>(emb1, w1, b1, nv1);
  nv_gemm_kernel<<<448, 256, 0, stream>>>(emb2, w2, b2, nv2);

  adj_gemm_kernel<<<896, 256, 0, stream>>>(nv1, nv2, init_adj, epoch, P);

  hist_kernel<<<dim3(N_VARS * N_VARS / (256 * 8), NUM_ADJ), 256, 0, stream>>>(P, hist);
  thresh_kernel<<<NUM_ADJ, 32, 0, stream>>>(hist, thr);
  finalize_kernel<<<dim3(N_VARS, 8), 256, 0, stream>>>(P, thr, time_indices, out);
}
